// Beam_Search_78039555768818
// MI455X (gfx1250) — compile-verified, matrix operand order checked by reading, not run on hardware
//
#include <hip/hip_runtime.h>

// ---------------- problem constants ----------------
static constexpr int S_ = 128, B_ = 64, E_ = 512, H_ = 1024, V_ = 32000;
static constexpr int BEAM_ = 2, MAXLEN_ = 32, EOS_ = 3, SOS_ = 2;
static constexpr int H3_ = 3 * H_;
static constexpr int BK_ = B_ * BEAM_;      // 128 decoder rows
static constexpr int DX_ = E_ + H_;         // 1536 decoder GRU input
static constexpr int HC_ = 2 * H_;          // 2048 [hn|ctx]

// ---------------- types ----------------
typedef __attribute__((ext_vector_type(16))) __bf16 bf16x16;
typedef __attribute__((ext_vector_type(8)))  float  f32x8;

struct Frag32B { uint4 lo; uint4 hi; };

__device__ __forceinline__ unsigned short f32_to_bf16(float f) {
  unsigned int u = __builtin_bit_cast(unsigned int, f);
  unsigned int r = u + 0x7FFFu + ((u >> 16) & 1u);
  return (unsigned short)(r >> 16);
}

__device__ __forceinline__ bf16x16 load_frag(const unsigned short* p0,
                                             const unsigned short* p1) {
  Frag32B r;
  r.lo = *reinterpret_cast<const uint4*>(p0);
  r.hi = *reinterpret_cast<const uint4*>(p1);
  return __builtin_bit_cast(bf16x16, r);
}

__device__ __forceinline__ float sigmoidf(float x) { return 1.0f / (1.0f + expf(-x)); }

// ---------------- weight convert: f32 [K,N] -> bf16 transposed [N,K] ----------------
__global__ void k_convertT(const float* __restrict__ W, unsigned short* __restrict__ Wt,
                           int K, int N) {
  size_t total = (size_t)K * (size_t)N;
  for (size_t i = (size_t)blockIdx.x * blockDim.x + threadIdx.x; i < total;
       i += (size_t)gridDim.x * blockDim.x) {
    size_t k = i / (size_t)N, n = i % (size_t)N;
    Wt[n * (size_t)K + k] = f32_to_bf16(W[i]);
  }
}

// ---------------- bf16 WMMA GEMM, N-strip per wave with MT m-tile accumulators ----
// C[MT*16, N] = A[MT*16, K] * Bt[N, K]^T (+bias). Each wave owns one 16-col strip
// and all MT m-tiles. Per k-step: load phase (B frag + MT A frags issued
// back-to-back -> one clause, partial waitcnts) then compute phase (MT WMMAs).
template <int MT>
__global__ void k_gemm_bf16_mt(const unsigned short* __restrict__ A,
                               const unsigned short* __restrict__ Bt,
                               const float* __restrict__ bias,
                               float* __restrict__ C,
                               int N, int K) {
  const int wavesPerBlock = blockDim.x >> 5;
  const int lane = threadIdx.x & 31;
  const int nt = N >> 4;
  const int hi8 = ((lane >> 4) & 1) << 3;          // 0 or 8
  const int l15 = lane & 15;

  const unsigned short* abase[MT];
#pragma unroll
  for (int mi = 0; mi < MT; ++mi)
    abase[mi] = A + (size_t)((mi << 4) + l15) * K;

  for (int tn = blockIdx.x * wavesPerBlock + (threadIdx.x >> 5); tn < nt;
       tn += gridDim.x * wavesPerBlock) {
    const unsigned short* brow = Bt + (size_t)((tn << 4) + l15) * K;
    f32x8 acc[MT];
#pragma unroll
    for (int mi = 0; mi < MT; ++mi)
      acc[mi] = (f32x8){0.f, 0.f, 0.f, 0.f, 0.f, 0.f, 0.f, 0.f};

    for (int k0 = 0; k0 < K; k0 += 32) {
      // ---- load phase: issue all fragments before any WMMA ----
      bf16x16 bfrag = load_frag(brow + k0 + hi8, brow + k0 + 16 + hi8);
      bf16x16 afrag[MT];
#pragma unroll
      for (int mi = 0; mi < MT; ++mi)
        afrag[mi] = load_frag(abase[mi] + k0 + hi8, abase[mi] + k0 + 16 + hi8);
      // ---- compute phase ----
#pragma unroll
      for (int mi = 0; mi < MT; ++mi)
        acc[mi] = __builtin_amdgcn_wmma_f32_16x16x32_bf16(false, afrag[mi], false, bfrag,
                                                          (short)0, acc[mi], false, false);
    }

    const int ccol  = (tn << 4) + l15;
    const int crow0 = (((lane >> 4) & 1) << 3);
    const float bv = bias ? bias[ccol] : 0.0f;
#pragma unroll
    for (int mi = 0; mi < MT; ++mi)
#pragma unroll
      for (int r = 0; r < 8; ++r)
        C[(size_t)((mi << 4) + crow0 + r) * N + ccol] = acc[mi][r] + bv;
  }
}

// ---------------- encoder embedding gather (f32 -> bf16) ----------------
__global__ void k_enc_embed(const int* __restrict__ ids, const float* __restrict__ emb,
                            unsigned short* __restrict__ xbf, int t) {
  int i = blockIdx.x * blockDim.x + threadIdx.x;     // B_*E_
  if (i >= B_ * E_) return;
  int b = i / E_, e = i % E_;
  int tok = ids[t * B_ + b];
  xbf[i] = f32_to_bf16(emb[(size_t)tok * E_ + e]);
}

// ---------------- GRU gate fusion ----------------
__global__ void k_gru_gate(const float* __restrict__ gx, const float* __restrict__ gh,
                           const float* __restrict__ bias, const float* __restrict__ hprev,
                           float* __restrict__ hf, unsigned short* __restrict__ hbf,
                           int M, int ldbf) {
  int i = blockIdx.x * blockDim.x + threadIdx.x;
  if (i >= M * H_) return;
  int m = i >> 10, j = i & (H_ - 1);
  const float* gxr = gx + (size_t)m * H3_;
  const float* ghr = gh + (size_t)m * H3_;
  float r = sigmoidf(gxr[j] + bias[j] + ghr[j]);
  float z = sigmoidf(gxr[H_ + j] + bias[H_ + j] + ghr[H_ + j]);
  float n = tanhf(gxr[2 * H_ + j] + bias[2 * H_ + j] + r * ghr[2 * H_ + j]);
  float hp = hprev ? hprev[i] : 0.0f;
  float h = (1.0f - z) * n + z * hp;
  hf[i] = h;
  hbf[(size_t)m * ldbf + j] = f32_to_bf16(h);
}

// ---------------- decoder init: last hidden, beam state ----------------
__global__ void k_dec_init(const float* __restrict__ enc_out, const int* __restrict__ input_len,
                           float* __restrict__ hdec, unsigned short* __restrict__ hbf,
                           int* __restrict__ tokens, float* __restrict__ scores,
                           float* __restrict__ probs, int* __restrict__ fin,
                           int* __restrict__ cur) {
  int b = blockIdx.x, tid = threadIdx.x;
  int last = input_len[b] - 1;
  for (int j = tid; j < H_; j += blockDim.x) {
    float v = enc_out[((size_t)last * B_ + b) * H_ + j];
    unsigned short bv = f32_to_bf16(v);
    for (int k = 0; k < BEAM_; ++k) {
      hdec[(size_t)(b * BEAM_ + k) * H_ + j] = v;
      hbf [(size_t)(b * BEAM_ + k) * H_ + j] = bv;
    }
  }
  if (tid < BEAM_) {
    int k = tid, row = b * BEAM_ + k;
    for (int j = 0; j < MAXLEN_; ++j) { tokens[row * MAXLEN_ + j] = EOS_; scores[row * MAXLEN_ + j] = 0.f; }
    tokens[row * MAXLEN_] = SOS_;
    scores[row * MAXLEN_] = 1.0f;
    probs[row] = (k == 0) ? 1.0f : 0.0f;
    fin[row] = 0;
    cur[row] = SOS_;
  }
}

// ---------------- decoder embedding into xcat[:, 0:E] ----------------
__global__ void k_dec_embed(const int* __restrict__ cur, const float* __restrict__ emb,
                            unsigned short* __restrict__ xcat) {
  int i = blockIdx.x * blockDim.x + threadIdx.x;     // BK_*E_
  if (i >= BK_ * E_) return;
  int row = i / E_, e = i % E_;
  xcat[(size_t)row * DX_ + e] = f32_to_bf16(emb[(size_t)cur[row] * E_ + e]);
}

// ---------------- attention: scores/softmax/context, write ctx + bf16 slices ----------------
__global__ void k_attention(const float* __restrict__ enc_out, const float* __restrict__ hdec,
                            const int* __restrict__ input_len, float* __restrict__ ctx,
                            unsigned short* __restrict__ xcat, unsigned short* __restrict__ hcat) {
  __shared__ float sh[H_];
  __shared__ float sc[S_];
  __shared__ float red[S_];
  int bk = blockIdx.x, b = bk >> 1, tid = threadIdx.x;   // 128 threads
  for (int j = tid; j < H_; j += S_) sh[j] = hdec[(size_t)bk * H_ + j];
  __syncthreads();
  const float* eb = enc_out + ((size_t)tid * B_ + b) * H_;
  float dot = 0.0f;
  for (int j = 0; j < H_; ++j) dot += eb[j] * sh[j];
  int len = input_len[b];
  float sv = (tid < len) ? dot : -1e9f;
  red[tid] = sv;
  __syncthreads();
  for (int off = 64; off > 0; off >>= 1) {
    if (tid < off) red[tid] = fmaxf(red[tid], red[tid + off]);
    __syncthreads();
  }
  float mx = red[0];
  __syncthreads();
  float ev = expf(sv - mx);
  red[tid] = ev;
  __syncthreads();
  for (int off = 64; off > 0; off >>= 1) {
    if (tid < off) red[tid] += red[tid + off];
    __syncthreads();
  }
  float inv = 1.0f / red[0];
  __syncthreads();
  sc[tid] = ev * inv;
  __syncthreads();
  for (int j = tid; j < H_; j += S_) {
    float acc = 0.0f;
    for (int s = 0; s < S_; ++s) acc += sc[s] * enc_out[((size_t)s * B_ + b) * H_ + j];
    ctx[(size_t)bk * H_ + j] = acc;
    unsigned short bv = f32_to_bf16(acc);
    xcat[(size_t)bk * DX_ + E_ + j] = bv;
    hcat[(size_t)bk * HC_ + H_ + j] = bv;
  }
}

// ---------------- softmax over V + candidate scores (in place) ----------------
__global__ void k_softmax_cand(float* __restrict__ logits, const float* __restrict__ probs,
                               const int* __restrict__ fin) {
  __shared__ float red[256];
  int row = blockIdx.x, tid = threadIdx.x;             // 256 threads
  float* l = logits + (size_t)row * V_;
  float m = -3.4e38f;
  for (int v = tid; v < V_; v += 256) m = fmaxf(m, l[v]);
  red[tid] = m; __syncthreads();
  for (int off = 128; off > 0; off >>= 1) {
    if (tid < off) red[tid] = fmaxf(red[tid], red[tid + off]);
    __syncthreads();
  }
  m = red[0]; __syncthreads();
  float s = 0.0f;
  for (int v = tid; v < V_; v += 256) s += expf(l[v] - m);
  red[tid] = s; __syncthreads();
  for (int off = 128; off > 0; off >>= 1) {
    if (tid < off) red[tid] += red[tid + off];
    __syncthreads();
  }
  s = red[0];
  float pr = probs[row];
  int f = fin[row];
  float invs = pr / s;
  for (int v = tid; v < V_; v += 256)
    l[v] = f ? ((v == EOS_) ? pr : 0.0f) : expf(l[v] - m) * invs;
}

// ---------------- top-2 over BEAM*V per batch ----------------
__global__ void k_top2(const float* __restrict__ cand, float* __restrict__ vals,
                       int* __restrict__ idx) {
  __shared__ float sv0[256], sv1[256];
  __shared__ int   si0[256], si1[256];
  int b = blockIdx.x, tid = threadIdx.x;               // 256 threads
  float v0 = -3.4e38f, v1 = -3.4e38f; int i0 = 0, i1 = 0;
  const int tot = BEAM_ * V_;
  for (int i = tid; i < tot; i += 256) {
    float v = cand[(size_t)(b * BEAM_ + i / V_) * V_ + (i % V_)];
    if (v > v0)      { v1 = v0; i1 = i0; v0 = v; i0 = i; }
    else if (v > v1) { v1 = v;  i1 = i; }
  }
  sv0[tid] = v0; si0[tid] = i0; sv1[tid] = v1; si1[tid] = i1;
  __syncthreads();
  for (int off = 128; off > 0; off >>= 1) {
    if (tid < off) {
      float a0 = sv0[tid], a1 = sv1[tid]; int j0 = si0[tid], j1 = si1[tid];
      float c[2] = { sv0[tid + off], sv1[tid + off] };
      int   ci[2] = { si0[tid + off], si1[tid + off] };
      for (int q = 0; q < 2; ++q) {
        if (c[q] > a0)      { a1 = a0; j1 = j0; a0 = c[q]; j0 = ci[q]; }
        else if (c[q] > a1) { a1 = c[q]; j1 = ci[q]; }
      }
      sv0[tid] = a0; si0[tid] = j0; sv1[tid] = a1; si1[tid] = j1;
    }
    __syncthreads();
  }
  if (tid == 0) {
    vals[b * BEAM_ + 0] = sv0[0]; idx[b * BEAM_ + 0] = si0[0];
    vals[b * BEAM_ + 1] = sv1[0]; idx[b * BEAM_ + 1] = si1[0];
  }
}

// ---------------- beam state update ----------------
__global__ void k_beam_update(const float* __restrict__ vals, const int* __restrict__ idx,
                              const float* __restrict__ hn, float* __restrict__ hdec,
                              unsigned short* __restrict__ hbf, int* __restrict__ tokens,
                              float* __restrict__ scores, float* __restrict__ probs,
                              int* __restrict__ fin, int* __restrict__ cur, int t) {
  int b = blockIdx.x, tid = threadIdx.x;               // 128 threads
  int bidx[BEAM_], tnew[BEAM_];
  for (int k = 0; k < BEAM_; ++k) {
    int ix = idx[b * BEAM_ + k];
    bidx[k] = ix / V_;
    tnew[k] = ix % V_;
  }
  for (int j = tid; j < BEAM_ * H_; j += blockDim.x) {
    int k = j / H_, jj = j % H_;
    float v = hn[(size_t)(b * BEAM_ + bidx[k]) * H_ + jj];
    hdec[(size_t)(b * BEAM_ + k) * H_ + jj] = v;
    hbf [(size_t)(b * BEAM_ + k) * H_ + jj] = f32_to_bf16(v);
  }
  if (tid == 0) {
    int   to_[BEAM_][MAXLEN_];
    float so_[BEAM_][MAXLEN_];
    int   fo_[BEAM_];
    for (int k = 0; k < BEAM_; ++k) {
      int row = b * BEAM_ + k;
      for (int j = 0; j < MAXLEN_; ++j) {
        to_[k][j] = tokens[row * MAXLEN_ + j];
        so_[k][j] = scores[row * MAXLEN_ + j];
      }
      fo_[k] = fin[row];
    }
    for (int k = 0; k < BEAM_; ++k) {
      int row = b * BEAM_ + k, bs = bidx[k];
      for (int j = 0; j < MAXLEN_; ++j) {
        tokens[row * MAXLEN_ + j] = to_[bs][j];
        scores[row * MAXLEN_ + j] = so_[bs][j];
      }
      int finp = fo_[bs];
      float v = vals[b * BEAM_ + k];
      tokens[row * MAXLEN_ + t] = finp ? EOS_ : tnew[k];
      scores[row * MAXLEN_ + t] = finp ? 0.0f : v;
      fin[row]  = finp | (tnew[k] == EOS_);
      probs[row] = v;
      cur[row]  = tnew[k];
    }
  }
}

// ---------------- pick best beam, write output ----------------
__global__ void k_finalize(const float* __restrict__ probs, const int* __restrict__ fin,
                           const int* __restrict__ tokens, const float* __restrict__ scores,
                           float* __restrict__ out) {
  int b = blockIdx.x, tid = threadIdx.x;
  float p0 = fin[b * BEAM_ + 0] ? probs[b * BEAM_ + 0] : -1.0f;
  float p1 = fin[b * BEAM_ + 1] ? probs[b * BEAM_ + 1] : -1.0f;
  int best = (p1 > p0) ? 1 : 0;
  int row = b * BEAM_ + best;
  for (int j = tid; j < MAXLEN_; j += blockDim.x) {
    out[b * MAXLEN_ + j]                 = (float)tokens[row * MAXLEN_ + j];
    out[B_ * MAXLEN_ + b * MAXLEN_ + j]  = scores[row * MAXLEN_ + j];
  }
}

// ---------------- host driver ----------------
extern "C" void kernel_launch(void* const* d_in, const int* in_sizes, int n_in,
                              void* d_out, int out_size, void* d_ws, size_t ws_size,
                              hipStream_t stream) {
  (void)in_sizes; (void)n_in; (void)out_size; (void)ws_size;
  const int*   input_ids = (const int*)  d_in[0];
  const int*   input_len = (const int*)  d_in[1];
  const float* enc_emb   = (const float*)d_in[2];
  const float* enc_Wx    = (const float*)d_in[3];
  const float* enc_Wh    = (const float*)d_in[4];
  const float* enc_b     = (const float*)d_in[5];
  const float* dec_emb   = (const float*)d_in[6];
  const float* dec_Wx    = (const float*)d_in[7];
  const float* dec_Wh    = (const float*)d_in[8];
  const float* dec_b     = (const float*)d_in[9];
  const float* Wo        = (const float*)d_in[10];
  const float* bo        = (const float*)d_in[11];
  float* out = (float*)d_out;

  char* w = (char*)d_ws;
  auto carve = [&](size_t bytes) {
    char* p = w;
    w += (bytes + 255) & ~(size_t)255;
    return p;
  };
  unsigned short* encWxT = (unsigned short*)carve((size_t)H3_ * E_  * 2);
  unsigned short* encWhT = (unsigned short*)carve((size_t)H3_ * H_  * 2);
  unsigned short* decWxT = (unsigned short*)carve((size_t)H3_ * DX_ * 2);
  unsigned short* decWhT = (unsigned short*)carve((size_t)H3_ * H_  * 2);
  unsigned short* WoT    = (unsigned short*)carve((size_t)V_  * HC_ * 2);
  float*          enc_out= (float*)carve((size_t)S_ * B_ * H_ * 4);
  unsigned short* xbf    = (unsigned short*)carve((size_t)B_  * E_  * 2);
  unsigned short* hbfE   = (unsigned short*)carve((size_t)B_  * H_  * 2);
  float*          gx     = (float*)carve((size_t)BK_ * H3_ * 4);
  float*          gh     = (float*)carve((size_t)BK_ * H3_ * 4);
  unsigned short* xcat   = (unsigned short*)carve((size_t)BK_ * DX_ * 2);
  unsigned short* hcat   = (unsigned short*)carve((size_t)BK_ * HC_ * 2);
  float*          ctx    = (float*)carve((size_t)BK_ * H_ * 4);
  float*          hn     = (float*)carve((size_t)BK_ * H_ * 4);
  float*          hdec   = (float*)carve((size_t)BK_ * H_ * 4);
  unsigned short* hbfD   = (unsigned short*)carve((size_t)BK_ * H_ * 2);
  float*          logits = (float*)carve((size_t)BK_ * V_ * 4);
  int*            tokens = (int*)  carve((size_t)BK_ * MAXLEN_ * 4);
  float*          scores = (float*)carve((size_t)BK_ * MAXLEN_ * 4);
  float*          probs  = (float*)carve((size_t)BK_ * 4);
  int*            fin    = (int*)  carve((size_t)BK_ * 4);
  int*            cur    = (int*)  carve((size_t)BK_ * 4);
  float*          vals   = (float*)carve((size_t)BK_ * 4);
  int*            idxb   = (int*)  carve((size_t)BK_ * 4);

  // --- one-time weight conversion (bf16, transposed so GEMM B-loads are contiguous) ---
  k_convertT<<<512,  256, 0, stream>>>(enc_Wx, encWxT, E_,  H3_);
  k_convertT<<<512,  256, 0, stream>>>(enc_Wh, encWhT, H_,  H3_);
  k_convertT<<<512,  256, 0, stream>>>(dec_Wx, decWxT, DX_, H3_);
  k_convertT<<<512,  256, 0, stream>>>(dec_Wh, decWhT, H_,  H3_);
  k_convertT<<<2048, 256, 0, stream>>>(Wo,     WoT,    HC_, V_);

  hipMemsetAsync(hbfE, 0, (size_t)B_ * H_ * 2, stream);   // h0 = 0 (bf16 zero == 0x0000)

  // --- encoder recurrence: 128 sequential steps; M=64 -> MT=4 accumulators/wave ---
  for (int t = 0; t < S_; ++t) {
    k_enc_embed<<<(B_ * E_ + 255) / 256, 256, 0, stream>>>(input_ids, enc_emb, xbf, t);
    {
      int blocks = (H3_ / 16 + 7) / 8;   // 24 blocks x 8 waves = 192 n-strips
      k_gemm_bf16_mt<4><<<blocks, 256, 0, stream>>>(xbf,  encWxT, nullptr, gx, H3_, E_);
      k_gemm_bf16_mt<4><<<blocks, 256, 0, stream>>>(hbfE, encWhT, nullptr, gh, H3_, H_);
    }
    const float* hprev = (t == 0) ? nullptr : (enc_out + (size_t)(t - 1) * B_ * H_);
    k_gru_gate<<<(B_ * H_ + 255) / 256, 256, 0, stream>>>(
        gx, gh, enc_b, hprev, enc_out + (size_t)t * B_ * H_, hbfE, B_, H_);
  }

  // --- decoder init ---
  k_dec_init<<<B_, 128, 0, stream>>>(enc_out, input_len, hdec, hbfD,
                                     tokens, scores, probs, fin, cur);

  // --- beam-search decode: 31 steps; M=128 -> MT=8 accumulators/wave ---
  for (int t = 1; t < MAXLEN_; ++t) {
    k_dec_embed<<<(BK_ * E_ + 255) / 256, 256, 0, stream>>>(cur, dec_emb, xcat);
    k_attention<<<BK_, 128, 0, stream>>>(enc_out, hdec, input_len, ctx, xcat, hcat);
    {
      int blocks = (H3_ / 16 + 7) / 8;
      k_gemm_bf16_mt<8><<<blocks, 256, 0, stream>>>(xcat, decWxT, nullptr, gx, H3_, DX_);
      k_gemm_bf16_mt<8><<<blocks, 256, 0, stream>>>(hbfD, decWhT, nullptr, gh, H3_, H_);
    }
    k_gru_gate<<<(BK_ * H_ + 255) / 256, 256, 0, stream>>>(
        gx, gh, dec_b, hdec, hn, hcat, BK_, HC_);      // hn bf16 into hcat[:, 0:H]
    {
      int blocks = (V_ / 16 + 7) / 8;                  // 250 blocks x 8 waves = 2000 n-strips
      k_gemm_bf16_mt<8><<<blocks, 256, 0, stream>>>(hcat, WoT, bo, logits, V_, HC_);
    }
    k_softmax_cand<<<BK_, 256, 0, stream>>>(logits, probs, fin);
    k_top2<<<B_, 256, 0, stream>>>(logits, vals, idxb);
    k_beam_update<<<B_, 128, 0, stream>>>(vals, idxb, hn, hdec, hbfD,
                                          tokens, scores, probs, fin, cur, t);
  }

  k_finalize<<<B_, 32, 0, stream>>>(probs, fin, tokens, scores, out);
}
